// RNNet_80341658239707
// MI455X (gfx1250) — compile-verified, matrix operand order checked by reading, not run on hardware
//
#include <hip/hip_runtime.h>

typedef __attribute__((ext_vector_type(16))) _Float16 v16h;
typedef __attribute__((ext_vector_type(8)))  _Float16 v8h;
typedef __attribute__((ext_vector_type(4)))  _Float16 v4h;
typedef __attribute__((ext_vector_type(8)))  float    v8f;

#define B_  4096
#define C_  9
#define T_  32
#define NB_ 32
#define IN_ 288
#define H_  128
#define G4_ 512   // 4*H

__device__ __forceinline__ float sigf(float x) {
    return 1.0f / (1.0f + __expf(-x));
}

// ---- CDNA5 async global->LDS (ASYNCcnt) helpers --------------------------
// LDS aperture places the byte offset in addr[31:0], so truncating a generic
// __shared__ pointer yields the wave-relative LDS byte address the async op
// wants in VDST.
__device__ __forceinline__ unsigned lds_off(const void* p) {
    return (unsigned)(unsigned long long)p;
}

__device__ __forceinline__ void async_b128(const void* gptr, const void* lptr) {
    unsigned l = lds_off(lptr);
    unsigned long long g = (unsigned long long)gptr;
    asm volatile("global_load_async_to_lds_b128 %0, %1, off"
                 :: "v"(l), "v"(g) : "memory");
}

__device__ __forceinline__ void wait_async0(void) {
#if __has_builtin(__builtin_amdgcn_s_wait_asynccnt)
    __builtin_amdgcn_s_wait_asynccnt(0);
#else
    asm volatile("s_wait_asynccnt 0" ::: "memory");
#endif
}

// ---------------------------------------------------------------------------
// Pre-pass: f32 -> f16 conversion (x, w_ih, w_hh). Vectorized 4-wide.
// ---------------------------------------------------------------------------
__global__ __launch_bounds__(256) void f32_to_f16(
    const float* __restrict__ s, _Float16* __restrict__ d, int nquads)
{
    int i = blockIdx.x * blockDim.x + threadIdx.x;
    const int stride = gridDim.x * blockDim.x;
    for (; i < nquads; i += stride) {
        const float4 v = ((const float4*)s)[i];
        v4h o = { (_Float16)v.x, (_Float16)v.y, (_Float16)v.z, (_Float16)v.w };
        ((v4h*)d)[i] = o;
    }
}

// ---------------------------------------------------------------------------
// Kernel 1: gx = x_seq @ w_ih^T + (b_ih+b_hh).  M=131072, N=512, K=288.
// f16 WMMA, fp32 accumulate. Block: 256 thr (8 waves); tile M=32 (one batch
// row b, all t), N=256. Tiles staged with double-buffered async b128 copies
// so the copy of K-chunk c+1 overlaps the WMMA of chunk c.
// ---------------------------------------------------------------------------
__global__ __launch_bounds__(256) void gemm1_gx(
    const _Float16* __restrict__ x16, const _Float16* __restrict__ wih16,
    const float* __restrict__ b_ih, const float* __restrict__ b_hh,
    _Float16* __restrict__ gx)
{
    __shared__ __attribute__((aligned(16))) _Float16 sA[2][32 * 32];    //  4 KB
    __shared__ __attribute__((aligned(16))) _Float16 sB[2][256 * 32];   // 32 KB

    const int tid   = threadIdx.x;
    const int lane  = tid & 31;
    const int wid   = tid >> 5;
    const int mhalf = wid >> 2;          // 0..1 -> 16-row half
    const int wq    = wid & 3;           // 0..3 -> 64-col quarter
    const int b     = blockIdx.x;        // rows b*32 .. b*32+31
    const int n0    = blockIdx.y * 256;  // 0 or 256
    const int l15   = lane & 15;
    const int kb    = (lane & 16) ? 8 : 0;
    const int hi8   = kb;

    v8f acc[4];
#pragma unroll
    for (int j = 0; j < 4; ++j) acc[j] = (v8f){0,0,0,0,0,0,0,0};

    // Stage K-chunk c into buffer `buf` (async; caller waits).
    auto stage = [&](int c, int buf) {
        // A: x16[b][c][0:32][0:32] is one contiguous 2KB block.
        if (tid < 128)
            async_b128(x16 + ((size_t)b * C_ + c) * (T_ * NB_) + tid * 8,
                       &sA[buf][tid * 8]);
        // B: w_ih row n0+tid, K columns c*32..c*32+31 (64B = 4 x b128).
        const _Float16* wsrc = wih16 + (size_t)(n0 + tid) * IN_ + c * 32;
#pragma unroll
        for (int j = 0; j < 4; ++j)
            async_b128(wsrc + j * 8, &sB[buf][tid * 32 + j * 8]);
    };

    stage(0, 0);
    wait_async0();
    __syncthreads();

    for (int c = 0; c < C_; ++c) {
        if (c + 1 < C_) stage(c + 1, (c + 1) & 1);

        const _Float16* sa = sA[c & 1];
        const _Float16* sb = sB[c & 1];

        // A fragment (16x32): row = mhalf*16 + (lane&15)
        const _Float16* ap = sa + (mhalf * 16 + l15) * 32;
        v8h alo = *(const v8h*)(ap + kb);
        v8h ahi = *(const v8h*)(ap + kb + 16);
        v16h a;
#pragma unroll
        for (int e = 0; e < 8; ++e) { a[e] = alo[e]; a[e + 8] = ahi[e]; }

#pragma unroll
        for (int j = 0; j < 4; ++j) {
            const _Float16* bp = sb + (wq * 64 + j * 16 + l15) * 32;
            v8h blo = *(const v8h*)(bp + kb);
            v8h bhi = *(const v8h*)(bp + kb + 16);
            v16h bf;
#pragma unroll
            for (int e = 0; e < 8; ++e) { bf[e] = blo[e]; bf[e + 8] = bhi[e]; }
            acc[j] = __builtin_amdgcn_wmma_f32_16x16x32_f16(
                false, a, false, bf, (short)0, acc[j], false, false);
        }
        wait_async0();
        __syncthreads();
    }

    // Bias + store f16 gx. C layout: elem v -> row v+hi8, col lane&15.
#pragma unroll
    for (int j = 0; j < 4; ++j) {
        const int n = n0 + wq * 64 + j * 16 + l15;
        const float bias = b_ih[n] + b_hh[n];
#pragma unroll
        for (int v = 0; v < 8; ++v) {
            const int m = b * 32 + mhalf * 16 + v + hi8;
            gx[(size_t)m * G4_ + n] = (_Float16)(acc[j][v] + bias);
        }
    }
}

// ---------------------------------------------------------------------------
// Kernel 2: LSTM recurrence. One block = 32 batch rows, fully independent
// (no grid sync needed). w_hh (512x128 f16 = 128KB) pulled into LDS once via
// async b128 copies; h tile in LDS; c state in registers. Each wave owns the
// SAME 32 columns of every gate region -> activation update is all in-lane.
// ---------------------------------------------------------------------------
__global__ __launch_bounds__(256) void lstm_rec(
    const _Float16* __restrict__ whh16, const _Float16* __restrict__ gx,
    _Float16* __restrict__ hout)
{
    __shared__ __attribute__((aligned(16))) _Float16 sW[G4_ * H_]; // 128 KB
    __shared__ __attribute__((aligned(16))) _Float16 sH[32 * H_];  // 8 KB

    const int tid   = threadIdx.x;
    const int lane  = tid & 31;
    const int wid   = tid >> 5;
    const int mhalf = wid >> 2;      // 0..1
    const int wq    = wid & 3;       // 0..3 -> 32-col slice of each gate
    const int b0    = blockIdx.x * 32;
    const int l15   = lane & 15;
    const int kb    = (lane & 16) ? 8 : 0;
    const int hi8   = kb;

    // One-shot async fill of w_hh: 128KB = 8192 b128 transfers.
    for (int i = tid; i < (G4_ * H_) / 8; i += 256)
        async_b128(whh16 + i * 8, &sW[i * 8]);
    for (int i = tid; i < 32 * H_; i += 256) sH[i] = (_Float16)0.0f;
    wait_async0();
    __syncthreads();

    float creg[2][8];
#pragma unroll
    for (int jt = 0; jt < 2; ++jt)
#pragma unroll
        for (int v = 0; v < 8; ++v) creg[jt][v] = 0.0f;

    for (int t = 0; t < T_; ++t) {
        v8f acc[4][2];
        // Init accumulators with precomputed gx (includes both biases).
#pragma unroll
        for (int g = 0; g < 4; ++g)
#pragma unroll
            for (int jt = 0; jt < 2; ++jt) {
                const int n = g * H_ + wq * 32 + jt * 16 + l15;
#pragma unroll
                for (int v = 0; v < 8; ++v) {
                    const int ml = mhalf * 16 + v + hi8;
                    const size_t row = (size_t)(b0 + ml) * T_ + t;
                    acc[g][jt][v] = (float)gx[row * G4_ + n];
                }
            }

        // gates += h @ w_hh^T  (K=128 as 4 wmma k-steps)
#pragma unroll
        for (int kt = 0; kt < 4; ++kt) {
            const _Float16* ap = sH + (mhalf * 16 + l15) * H_ + kt * 32;
            v8h alo = *(const v8h*)(ap + kb);
            v8h ahi = *(const v8h*)(ap + kb + 16);
            v16h a;
#pragma unroll
            for (int e = 0; e < 8; ++e) { a[e] = alo[e]; a[e + 8] = ahi[e]; }
#pragma unroll
            for (int g = 0; g < 4; ++g)
#pragma unroll
                for (int jt = 0; jt < 2; ++jt) {
                    const _Float16* bp =
                        sW + (g * H_ + wq * 32 + jt * 16 + l15) * H_ + kt * 32;
                    v8h blo = *(const v8h*)(bp + kb);
                    v8h bhi = *(const v8h*)(bp + kb + 16);
                    v16h bf;
#pragma unroll
                    for (int e = 0; e < 8; ++e) { bf[e] = blo[e]; bf[e + 8] = bhi[e]; }
                    acc[g][jt] = __builtin_amdgcn_wmma_f32_16x16x32_f16(
                        false, a, false, bf, (short)0, acc[g][jt], false, false);
                }
        }
        __syncthreads();   // everyone done reading sH for this step

        // Activations + state update, all in-lane.
#pragma unroll
        for (int jt = 0; jt < 2; ++jt) {
            const int j = wq * 32 + jt * 16 + l15;
#pragma unroll
            for (int v = 0; v < 8; ++v) {
                const float ig = sigf(acc[0][jt][v]);
                const float fg = sigf(acc[1][jt][v]);
                const float gg = tanhf(acc[2][jt][v]);
                const float og = sigf(acc[3][jt][v]);
                const float c  = fg * creg[jt][v] + ig * gg;
                creg[jt][v] = c;
                const float h = og * tanhf(c);
                const int m = mhalf * 16 + v + hi8;
                sH[m * H_ + j] = (_Float16)h;
            }
        }
        __syncthreads();   // new h visible before next step's reads
    }

    for (int i = tid; i < 32 * H_; i += 256)
        hout[(size_t)b0 * H_ + i] = sH[i];
}

// ---------------------------------------------------------------------------
// Dense tail: no nonlinearity between the two linear layers, so
// out = h @ (w2@w1)^T + (w2@b1 + b2) collapses to one 128-dot per row.
// ---------------------------------------------------------------------------
__global__ void dense_collapse(const float* __restrict__ w1, const float* __restrict__ b1,
                               const float* __restrict__ w2, const float* __restrict__ b2,
                               float* __restrict__ weff, float* __restrict__ beff)
{
    const int k = threadIdx.x; // 0..127
    float s = 0.0f;
    for (int n = 0; n < 256; ++n) s += w2[n] * w1[n * H_ + k];
    weff[k] = s;
    if (k == 0) {
        float bb = b2[0];
        for (int n = 0; n < 256; ++n) bb += w2[n] * b1[n];
        beff[0] = bb;
    }
}

__global__ __launch_bounds__(256) void dense_out(
    const _Float16* __restrict__ h, const float* __restrict__ weff,
    const float* __restrict__ beff, float* __restrict__ out)
{
    const int lane = threadIdx.x & 31;
    const int wid  = threadIdx.x >> 5;
    const int row  = blockIdx.x * 8 + wid;
    const _Float16* hr = h + (size_t)row * H_;
    float s = 0.0f;
#pragma unroll
    for (int i = 0; i < 4; ++i)
        s += (float)hr[lane + i * 32] * weff[lane + i * 32];
#pragma unroll
    for (int off = 16; off > 0; off >>= 1) s += __shfl_xor(s, off, 32);
    if (lane == 0) out[row] = s + beff[0];
}

// ---------------------------------------------------------------------------
extern "C" void kernel_launch(void* const* d_in, const int* in_sizes, int n_in,
                              void* d_out, int out_size, void* d_ws, size_t ws_size,
                              hipStream_t stream)
{
    const float* x    = (const float*)d_in[0];
    const float* w_ih = (const float*)d_in[1];
    const float* w_hh = (const float*)d_in[2];
    const float* b_ih = (const float*)d_in[3];
    const float* b_hh = (const float*)d_in[4];
    const float* w1   = (const float*)d_in[5];
    const float* b1   = (const float*)d_in[6];
    const float* w2   = (const float*)d_in[7];
    const float* b2   = (const float*)d_in[8];
    float* out = (float*)d_out;

    char* ws = (char*)d_ws;
    const size_t GX_BYTES  = (size_t)B_ * T_ * G4_ * 2;   // 134,217,728
    const size_t H_BYTES   = (size_t)B_ * H_ * 2;         //   1,048,576
    const size_t X16_BYTES = (size_t)B_ * C_ * T_ * NB_ * 2; // 75,497,472
    const size_t WIH_BYTES = (size_t)G4_ * IN_ * 2;       //     294,912
    const size_t WHH_BYTES = (size_t)G4_ * H_ * 2;        //     131,072

    size_t off = 0;
    _Float16* gx    = (_Float16*)(ws + off); off += GX_BYTES;
    _Float16* hh    = (_Float16*)(ws + off); off += H_BYTES;
    _Float16* x16   = (_Float16*)(ws + off); off += X16_BYTES;
    _Float16* wih16 = (_Float16*)(ws + off); off += WIH_BYTES;
    _Float16* whh16 = (_Float16*)(ws + off); off += WHH_BYTES;
    float*    weff  = (float*)(ws + off);    off += 128 * sizeof(float);
    float*    beff  = (float*)(ws + off);

    // f32 -> f16 pre-pass (independent of each other)
    f32_to_f16<<<2048, 256, 0, stream>>>(x, x16, (B_ * C_ * T_ * NB_) / 4);
    f32_to_f16<<<64,   256, 0, stream>>>(w_ih, wih16, (G4_ * IN_) / 4);
    f32_to_f16<<<32,   256, 0, stream>>>(w_hh, whh16, (G4_ * H_) / 4);

    dim3 g1(B_, 2);
    gemm1_gx<<<g1, 256, 0, stream>>>(x16, wih16, b_ih, b_hh, gx);
    lstm_rec<<<B_ / 32, 256, 0, stream>>>(whh16, gx, hh);
    dense_collapse<<<1, 128, 0, stream>>>(w1, b1, w2, b2, weff, beff);
    dense_out<<<B_ / 8, 256, 0, stream>>>(hh, weff, beff, out);
}
